// _PairBiasTransformerLayer_23819888623920
// MI455X (gfx1250) — compile-verified, matrix operand order checked by reading, not run on hardware
//
#include <hip/hip_runtime.h>

#define Bdim 2
#define Ldim 256
#define Hdim 512
#define Pdim 128
#define NH   8
#define FFdim 2048
#define LN_EPS 1e-5f

typedef __bf16 bf16_t;
typedef __attribute__((ext_vector_type(16))) __bf16 v16bf;
typedef __attribute__((ext_vector_type(8)))  __bf16 v8bf;
typedef __attribute__((ext_vector_type(8)))  float  v8f;

__device__ __forceinline__ float wave_sum(float v) {
#pragma unroll
  for (int o = 16; o > 0; o >>= 1) v += __shfl_xor(v, o, 32);
  return v;
}
__device__ __forceinline__ float wave_max(float v) {
#pragma unroll
  for (int o = 16; o > 0; o >>= 1) v = fmaxf(v, __shfl_xor(v, o, 32));
  return v;
}
__device__ __forceinline__ float gelu_f(float x) {
  return 0.5f * x * (1.0f + erff(x * 0.70710678118654752f));
}

// Async copy 16B global -> LDS (per lane). ISA §15.18 GLOBAL_LOAD_ASYNC_TO_LDS_B128,
// tracked by ASYNCcnt. Low 32 bits of a flat LDS pointer are the LDS byte offset
// (ISA §10.2 aperture table), which is what VDST wants.
__device__ __forceinline__ void async_load_b128(const void* gsrc, void* ldst) {
  unsigned l = (unsigned)(size_t)ldst;
  asm volatile("global_load_async_to_lds_b128 %0, %1, off"
               :: "v"(l), "v"(gsrc) : "memory");
}
__device__ __forceinline__ void wait_async() {
  asm volatile("s_wait_asynccnt 0x0" ::: "memory");
}

// Load one 16x32 bf16 WMMA operand fragment (A row-major, or B supplied as
// B^T row-major so its "rows" are K-contiguous). Per ISA 7.12.2: lanes 0-15
// hold K 0..7 (VGPR0-3) and 16..23 (VGPR4-7); lanes 16-31 hold the +8 halves.
// Each half is 8 contiguous bf16 = one 16-byte load.
__device__ __forceinline__ v16bf load_frag(const bf16_t* base, int stride, int lane) {
  const int m = lane & 15, half = lane >> 4;
  const bf16_t* r = base + (long)m * stride + half * 8;
  v8bf lo = *(const v8bf*)r;
  v8bf hi = *(const v8bf*)(r + 16);
  return __builtin_shufflevector(lo, hi, 0,1,2,3,4,5,6,7,8,9,10,11,12,13,14,15);
}

__device__ __forceinline__ v8f wmma_bf16(v16bf a, v16bf b, v8f c) {
  return __builtin_amdgcn_wmma_f32_16x16x32_bf16(false, a, false, b, (short)0, c,
                                                 false, false);
}

// ---------------- weight transpose + f32->bf16 ----------------
__global__ __launch_bounds__(256) void k_wt(const float* __restrict__ W,
                                            bf16_t* __restrict__ WT, int K, int N) {
  long idx = (long)blockIdx.x * 256 + threadIdx.x;
  if (idx >= (long)K * N) return;
  int k = (int)(idx / N), n = (int)(idx % N);
  WT[(long)n * K + k] = (bf16_t)W[idx];
}

// ---------------- LayerNorm over last dim D -> bf16 ----------------
__global__ __launch_bounds__(256) void k_ln_bf16(const float* __restrict__ x,
                                                 const float* __restrict__ g,
                                                 const float* __restrict__ b,
                                                 bf16_t* __restrict__ y, int D) {
  __shared__ float sa[8], sb[8], stot[2];
  const int tid = threadIdx.x, lane = tid & 31, wid = tid >> 5;
  const long row = blockIdx.x;
  const float* xr = x + row * D;
  float s = 0.f, s2 = 0.f;
  for (int i = tid; i < D; i += 256) { float v = xr[i]; s += v; s2 += v * v; }
  s = wave_sum(s); s2 = wave_sum(s2);
  if (lane == 0) { sa[wid] = s; sb[wid] = s2; }
  __syncthreads();
  if (tid == 0) {
    float a = 0.f, c = 0.f;
    for (int w = 0; w < 8; ++w) { a += sa[w]; c += sb[w]; }
    stot[0] = a; stot[1] = c;
  }
  __syncthreads();
  const float mean = stot[0] / D;
  const float rstd = rsqrtf(stot[1] / D - mean * mean + LN_EPS);
  for (int i = tid; i < D; i += 256)
    y[row * D + i] = (bf16_t)((xr[i] - mean) * rstd * g[i] + b[i]);
}

// ---------------- pair LN + bias projection: bias[b,h,i,j] ----------------
__global__ __launch_bounds__(256) void k_pair_bias(const float* __restrict__ pair,
                                                   const float* __restrict__ g,
                                                   const float* __restrict__ be,
                                                   const float* __restrict__ Wb,
                                                   float* __restrict__ biasOut) {
  __shared__ float sWb[Pdim * NH], sg[Pdim], sb[Pdim];
  const int tid = threadIdx.x, lane = tid & 31, wid = tid >> 5;
  for (int i = tid; i < Pdim * NH; i += 256) sWb[i] = Wb[i];
  for (int i = tid; i < Pdim; i += 256) { sg[i] = g[i]; sb[i] = be[i]; }
  __syncthreads();
  const long row = (long)blockIdx.x * 8 + wid;   // flat (b,i,j)
  const float* pr = pair + row * Pdim;
  const int p4 = lane * 4;
  float4 v = *(const float4*)(pr + p4);
  float s  = wave_sum(v.x + v.y + v.z + v.w);
  float s2 = wave_sum(v.x*v.x + v.y*v.y + v.z*v.z + v.w*v.w);
  const float mean = s * (1.0f / Pdim);
  const float rstd = rsqrtf(s2 * (1.0f / Pdim) - mean * mean + LN_EPS);
  float vv[4] = {v.x, v.y, v.z, v.w};
  float acc[NH];
#pragma unroll
  for (int h = 0; h < NH; ++h) acc[h] = 0.f;
#pragma unroll
  for (int e = 0; e < 4; ++e) {
    const int p = p4 + e;
    const float nv = (vv[e] - mean) * rstd * sg[p] + sb[p];
#pragma unroll
    for (int h = 0; h < NH; ++h) acc[h] += nv * sWb[p * NH + h];
  }
#pragma unroll
  for (int h = 0; h < NH; ++h) acc[h] = wave_sum(acc[h]);
  const int bb = (int)(row / ((long)Ldim * Ldim));
  const long rem = row - (long)bb * Ldim * Ldim;
  const int ii = (int)(rem / Ldim), jj = (int)(rem % Ldim);
#pragma unroll
  for (int h = 0; h < NH; ++h)
    if (lane == h)
      biasOut[(((long)bb * NH + h) * Ldim + ii) * Ldim + jj] = acc[h];
}

// ---------------- softmax over rows of 256 -> bf16 ----------------
__global__ __launch_bounds__(256) void k_softmax_bf16(const float* __restrict__ s,
                                                      bf16_t* __restrict__ attn) {
  const int lane = threadIdx.x & 31, wid = threadIdx.x >> 5;
  const long row = (long)blockIdx.x * 8 + wid;
  const float* sr = s + row * Ldim;
  float4 a0 = *(const float4*)(sr + lane * 8);
  float4 a1 = *(const float4*)(sr + lane * 8 + 4);
  float x[8] = {a0.x, a0.y, a0.z, a0.w, a1.x, a1.y, a1.z, a1.w};
  float m = x[0];
#pragma unroll
  for (int t = 1; t < 8; ++t) m = fmaxf(m, x[t]);
  m = wave_max(m);
  float sum = 0.f;
#pragma unroll
  for (int t = 0; t < 8; ++t) { x[t] = __expf(x[t] - m); sum += x[t]; }
  sum = wave_sum(sum);
  const float inv = 1.0f / sum;
  v8bf o;
#pragma unroll
  for (int t = 0; t < 8; ++t) o[t] = (bf16_t)(x[t] * inv);
  *(v8bf*)(attn + row * Ldim + lane * 8) = o;
}

// ---------------- generic strided-batched bf16 WMMA GEMM ----------------
// C[z] = epilogue( scale * A[z](MxK) @ Bt[z](NxK)^T ), block tile 128x64, BK=32.
// 256 threads = 8 waves in a 4x2 grid, 32x32 per wave (2x2 WMMA 16x16x32).
// Tiles staged with GLOBAL_LOAD_ASYNC_TO_LDS_B128 (ASYNCcnt), no VGPR round-trip.
template <int BKN>
__global__ __launch_bounds__(256) void k_gemm(
    const bf16_t* __restrict__ A, const bf16_t* __restrict__ Bm,
    float* __restrict__ Cf, bf16_t* __restrict__ Cbf,
    const float* __restrict__ resid, const float* __restrict__ biasN,
    int lda, int ldb, int ldc, int ldr,
    long sAo, long sAi, long sBo, long sBi,
    long sCo, long sCi, long sRo, long sRi, int zInner,
    int K, float scale, int doGelu) {
  __shared__ __align__(16) bf16_t As[128][32];
  __shared__ __align__(16) bf16_t Bs[64][32];
  const int tid = threadIdx.x, lane = tid & 31, wid = tid >> 5;
  const int z = blockIdx.z;
  const int zo = z / zInner, zi = z - zo * zInner;
  const bf16_t* Ab = A + (long)zo * sAo + (long)zi * sAi;
  const bf16_t* Bb = Bm + (long)zo * sBo + (long)zi * sBi;
  const long coff = (long)zo * sCo + (long)zi * sCi;
  const long roff = (long)zo * sRo + (long)zi * sRi;
  const int bm = blockIdx.y * 128, bn = blockIdx.x * 64;
  const int wm = (wid >> 1) * 32, wn = (wid & 1) * 32;
  v8f acc00 = {}, acc01 = {}, acc10 = {}, acc11 = {};
  const int arow = tid >> 1, akq = (tid & 1) * 16;
  const int brow = tid >> 2, bkq = (tid & 3) * 8;
  for (int k0 = 0; k0 < K; k0 += 32) {
    const bf16_t* asrc = Ab + (long)(bm + arow) * lda + k0 + akq;
    __builtin_prefetch(asrc + 32, 0, 0);       // next K-tile -> global_prefetch_b8
    async_load_b128(asrc, &As[arow][akq]);
    async_load_b128(asrc + 8, &As[arow][akq + 8]);
    if (BKN == 0) {                             // Bt given as [N][K] row-major
      async_load_b128(Bb + (long)(bn + brow) * ldb + k0 + bkq, &Bs[brow][bkq]);
    } else {                                    // B given as [K][N]; transpose
#pragma unroll
      for (int e = 0; e < 8; ++e) {
        const int idx = tid * 8 + e;
        const int kk = idx >> 6, nn = idx & 63;
        Bs[nn][kk] = Bb[(long)(k0 + kk) * ldb + bn + nn];
      }
    }
    wait_async();                               // our LDS writes landed
    __syncthreads();
    v16bf a0 = load_frag(&As[wm][0], 32, lane);
    v16bf a1 = load_frag(&As[wm + 16][0], 32, lane);
    v16bf b0 = load_frag(&Bs[wn][0], 32, lane);
    v16bf b1 = load_frag(&Bs[wn + 16][0], 32, lane);
    acc00 = wmma_bf16(a0, b0, acc00);
    acc01 = wmma_bf16(a0, b1, acc01);
    acc10 = wmma_bf16(a1, b0, acc10);
    acc11 = wmma_bf16(a1, b1, acc11);
    __syncthreads();
  }
  const int nl = lane & 15, mg = lane >> 4;
#pragma unroll
  for (int mi = 0; mi < 2; ++mi)
#pragma unroll
    for (int ni = 0; ni < 2; ++ni) {
      v8f accv = (mi == 0) ? (ni == 0 ? acc00 : acc01)
                           : (ni == 0 ? acc10 : acc11);
      const int rb = bm + wm + mi * 16 + mg * 8;
      const int col = bn + wn + ni * 16 + nl;
      const float bv = biasN ? biasN[col] : 0.0f;
#pragma unroll
      for (int r = 0; r < 8; ++r) {
        const int row = rb + r;
        float v = accv[r] * scale + bv;
        if (resid) v += resid[roff + (long)row * ldr + col];
        if (doGelu) v = gelu_f(v);
        const long o = coff + (long)row * ldc + col;
        if (Cf)  Cf[o] = v;
        if (Cbf) Cbf[o] = (bf16_t)v;
      }
    }
}

// ---------------- fused pair transition ----------------
// One block per (b, i, 128-wide j tile): LN(a_i+b_j+ob) -> LDS(bf16),
// H = gelu(Y@W1+b1) -> LDS(bf16), out = pair + H@W2 + b2. Keeps the two
// (B,L,L,P) intermediates out of HBM (saves ~134MB of traffic).
__global__ __launch_bounds__(256) void k_pair_update(
    const float* __restrict__ pair_in, const float* __restrict__ aI,
    const float* __restrict__ bJ, const float* __restrict__ ob,
    const float* __restrict__ g, const float* __restrict__ be,
    const bf16_t* __restrict__ W1t, const float* __restrict__ b1,
    const bf16_t* __restrict__ W2t, const float* __restrict__ b2,
    float* __restrict__ pair_out) {
  __shared__ __align__(16) bf16_t Y[128][128];
  __shared__ __align__(16) bf16_t Hs[128][128];
  __shared__ float sAi[Pdim], sG[Pdim], sBe[Pdim], sB1[Pdim], sB2[Pdim];
  const int tid = threadIdx.x, lane = tid & 31, wid = tid >> 5;
  const int b = blockIdx.z, irow = blockIdx.y, jt = blockIdx.x * 128;
  for (int p = tid; p < Pdim; p += 256) {
    sAi[p] = aI[((long)b * Ldim + irow) * Pdim + p] + ob[p];
    sG[p] = g[p]; sBe[p] = be[p]; sB1[p] = b1[p]; sB2[p] = b2[p];
  }
  __syncthreads();
  const int p4 = lane * 4;
  for (int jr = wid; jr < 128; jr += 8) {
    const float* br = bJ + ((long)b * Ldim + jt + jr) * Pdim + p4;
    float4 bv = *(const float4*)br;
    const float v0 = sAi[p4 + 0] + bv.x, v1 = sAi[p4 + 1] + bv.y;
    const float v2 = sAi[p4 + 2] + bv.z, v3 = sAi[p4 + 3] + bv.w;
    const float s  = wave_sum(v0 + v1 + v2 + v3);
    const float s2 = wave_sum(v0*v0 + v1*v1 + v2*v2 + v3*v3);
    const float mean = s * (1.0f / Pdim);
    const float rstd = rsqrtf(s2 * (1.0f / Pdim) - mean * mean + LN_EPS);
    Y[jr][p4 + 0] = (bf16_t)((v0 - mean) * rstd * sG[p4 + 0] + sBe[p4 + 0]);
    Y[jr][p4 + 1] = (bf16_t)((v1 - mean) * rstd * sG[p4 + 1] + sBe[p4 + 1]);
    Y[jr][p4 + 2] = (bf16_t)((v2 - mean) * rstd * sG[p4 + 2] + sBe[p4 + 2]);
    Y[jr][p4 + 3] = (bf16_t)((v3 - mean) * rstd * sG[p4 + 3] + sBe[p4 + 3]);
  }
  __syncthreads();
  const int wm = (wid >> 1) * 32, wn = (wid & 1) * 64;
  const int nl = lane & 15, mg = lane >> 4;
  {
    v8f acc[2][4] = {};
    for (int k0 = 0; k0 < Pdim; k0 += 32) {
      v16bf a0 = load_frag(&Y[wm][k0], 128, lane);
      v16bf a1 = load_frag(&Y[wm + 16][k0], 128, lane);
#pragma unroll
      for (int ni = 0; ni < 4; ++ni) {
        v16bf bw = load_frag(W1t + (long)(wn + ni * 16) * Pdim + k0, Pdim, lane);
        acc[0][ni] = wmma_bf16(a0, bw, acc[0][ni]);
        acc[1][ni] = wmma_bf16(a1, bw, acc[1][ni]);
      }
    }
#pragma unroll
    for (int mi = 0; mi < 2; ++mi)
#pragma unroll
      for (int ni = 0; ni < 4; ++ni) {
        const int rb = wm + mi * 16 + mg * 8;
        const int col = wn + ni * 16 + nl;
        const float bb = sB1[col];
#pragma unroll
        for (int r = 0; r < 8; ++r)
          Hs[rb + r][col] = (bf16_t)gelu_f(acc[mi][ni][r] + bb);
      }
  }
  __syncthreads();
  {
    v8f acc[2][4] = {};
    for (int k0 = 0; k0 < Pdim; k0 += 32) {
      v16bf a0 = load_frag(&Hs[wm][k0], 128, lane);
      v16bf a1 = load_frag(&Hs[wm + 16][k0], 128, lane);
#pragma unroll
      for (int ni = 0; ni < 4; ++ni) {
        v16bf bw = load_frag(W2t + (long)(wn + ni * 16) * Pdim + k0, Pdim, lane);
        acc[0][ni] = wmma_bf16(a0, bw, acc[0][ni]);
        acc[1][ni] = wmma_bf16(a1, bw, acc[1][ni]);
      }
    }
#pragma unroll
    for (int mi = 0; mi < 2; ++mi)
#pragma unroll
      for (int ni = 0; ni < 4; ++ni) {
        const int col = wn + ni * 16 + nl;
        const float bb = sB2[col];
#pragma unroll
        for (int r = 0; r < 8; ++r) {
          const int jrow = wm + mi * 16 + mg * 8 + r;
          const long o =
              (((long)b * Ldim + irow) * Ldim + (jt + jrow)) * Pdim + col;
          pair_out[o] = pair_in[o] + acc[mi][ni][r] + bb;
        }
      }
  }
}

extern "C" void kernel_launch(void* const* d_in, const int* in_sizes, int n_in,
                              void* d_out, int out_size, void* d_ws, size_t ws_size,
                              hipStream_t stream) {
  (void)in_sizes; (void)n_in; (void)out_size; (void)ws_size;
  const float* seq       = (const float*)d_in[0];
  const float* pair      = (const float*)d_in[1];
  const float* ln_seq_g  = (const float*)d_in[2];
  const float* ln_seq_b  = (const float*)d_in[3];
  const float* ln_pair_g = (const float*)d_in[4];
  const float* ln_pair_b = (const float*)d_in[5];
  const float* W_bias    = (const float*)d_in[6];
  const float* Wq        = (const float*)d_in[7];
  const float* Wk        = (const float*)d_in[8];
  const float* Wv        = (const float*)d_in[9];
  const float* Wo        = (const float*)d_in[10];
  const float* bo        = (const float*)d_in[11];
  const float* ff_ln_g   = (const float*)d_in[12];
  const float* ff_ln_b   = (const float*)d_in[13];
  const float* ff_W1     = (const float*)d_in[14];
  const float* ff_b1     = (const float*)d_in[15];
  const float* ff_W2     = (const float*)d_in[16];
  const float* ff_b2     = (const float*)d_in[17];
  const float* outer_Wi  = (const float*)d_in[18];
  const float* outer_Wj  = (const float*)d_in[19];
  const float* outer_b   = (const float*)d_in[20];
  const float* pu_ln_g   = (const float*)d_in[21];
  const float* pu_ln_b   = (const float*)d_in[22];
  const float* pu_W1     = (const float*)d_in[23];
  const float* pu_b1     = (const float*)d_in[24];
  const float* pu_W2     = (const float*)d_in[25];
  const float* pu_b2     = (const float*)d_in[26];

  float* out_seq  = (float*)d_out;
  float* out_pair = out_seq + (long)Bdim * Ldim * Hdim;

  char* wp = (char*)d_ws;
  auto alloc = [&](size_t bytes) -> void* {
    void* p = (void*)wp;
    wp += (bytes + 255) & ~(size_t)255;
    return p;
  };
  bf16_t* WqT   = (bf16_t*)alloc((size_t)Hdim * Hdim * 2);
  bf16_t* WkT   = (bf16_t*)alloc((size_t)Hdim * Hdim * 2);
  bf16_t* WvT   = (bf16_t*)alloc((size_t)Hdim * Hdim * 2);
  bf16_t* WoT   = (bf16_t*)alloc((size_t)Hdim * Hdim * 2);
  bf16_t* W1T   = (bf16_t*)alloc((size_t)Hdim * FFdim * 2);
  bf16_t* W2T   = (bf16_t*)alloc((size_t)FFdim * Hdim * 2);
  bf16_t* WoiT  = (bf16_t*)alloc((size_t)Hdim * Pdim * 2);
  bf16_t* WojT  = (bf16_t*)alloc((size_t)Hdim * Pdim * 2);
  bf16_t* Wp1T  = (bf16_t*)alloc((size_t)Pdim * Pdim * 2);
  bf16_t* Wp2T  = (bf16_t*)alloc((size_t)Pdim * Pdim * 2);
  bf16_t* seqnB = (bf16_t*)alloc((size_t)Bdim * Ldim * Hdim * 2);
  bf16_t* Qb    = (bf16_t*)alloc((size_t)Bdim * Ldim * Hdim * 2);
  bf16_t* Kb    = (bf16_t*)alloc((size_t)Bdim * Ldim * Hdim * 2);
  bf16_t* Vb    = (bf16_t*)alloc((size_t)Bdim * Ldim * Hdim * 2);
  float*  scor  = (float*) alloc((size_t)Bdim * NH * Ldim * Ldim * 4);
  bf16_t* attnB = (bf16_t*)alloc((size_t)Bdim * NH * Ldim * Ldim * 2);
  bf16_t* aoB   = (bf16_t*)alloc((size_t)Bdim * Ldim * Hdim * 2);
  float*  seq2  = (float*) alloc((size_t)Bdim * Ldim * Hdim * 4);
  bf16_t* xlnB  = (bf16_t*)alloc((size_t)Bdim * Ldim * Hdim * 2);
  bf16_t* ffhB  = (bf16_t*)alloc((size_t)Bdim * Ldim * FFdim * 2);
  bf16_t* soutB = (bf16_t*)alloc((size_t)Bdim * Ldim * Hdim * 2);
  float*  aF    = (float*) alloc((size_t)Bdim * Ldim * Pdim * 4);
  float*  bF    = (float*) alloc((size_t)Bdim * Ldim * Pdim * 4);

  auto wt = [&](const float* W, bf16_t* WT, int K, int N) {
    long n = (long)K * N;
    k_wt<<<dim3((unsigned)((n + 255) / 256)), 256, 0, stream>>>(W, WT, K, N);
  };
  wt(Wq, WqT, Hdim, Hdim);   wt(Wk, WkT, Hdim, Hdim);
  wt(Wv, WvT, Hdim, Hdim);   wt(Wo, WoT, Hdim, Hdim);
  wt(ff_W1, W1T, Hdim, FFdim); wt(ff_W2, W2T, FFdim, Hdim);
  wt(outer_Wi, WoiT, Hdim, Pdim); wt(outer_Wj, WojT, Hdim, Pdim);
  wt(pu_W1, Wp1T, Pdim, Pdim); wt(pu_W2, Wp2T, Pdim, Pdim);

  const int Mtok = Bdim * Ldim;  // 512
  k_ln_bf16<<<Mtok, 256, 0, stream>>>(seq, ln_seq_g, ln_seq_b, seqnB, Hdim);
  k_pair_bias<<<(Bdim * Ldim * Ldim) / 8, 256, 0, stream>>>(
      pair, ln_pair_g, ln_pair_b, W_bias, scor);

  const long LH = (long)Ldim * Hdim;      // 131072
  const long LL = (long)Ldim * Ldim;      // 65536
  // Q/K/V projections (M=512,N=512,K=512)
  k_gemm<0><<<dim3(Hdim / 64, Mtok / 128, 1), 256, 0, stream>>>(
      seqnB, WqT, nullptr, Qb, nullptr, nullptr, Hdim, Hdim, Hdim, 0,
      0, 0, 0, 0, 0, 0, 0, 0, 1, Hdim, 1.0f, 0);
  k_gemm<0><<<dim3(Hdim / 64, Mtok / 128, 1), 256, 0, stream>>>(
      seqnB, WkT, nullptr, Kb, nullptr, nullptr, Hdim, Hdim, Hdim, 0,
      0, 0, 0, 0, 0, 0, 0, 0, 1, Hdim, 1.0f, 0);
  k_gemm<0><<<dim3(Hdim / 64, Mtok / 128, 1), 256, 0, stream>>>(
      seqnB, WvT, nullptr, Vb, nullptr, nullptr, Hdim, Hdim, Hdim, 0,
      0, 0, 0, 0, 0, 0, 0, 0, 1, Hdim, 1.0f, 0);
  // scores = QK^T * hd^-.5 + bias (resid=bias tensor, in-place over scor)
  k_gemm<0><<<dim3(Ldim / 64, Ldim / 128, Bdim * NH), 256, 0, stream>>>(
      Qb, Kb, scor, nullptr, scor, nullptr, Hdim, Hdim, Ldim, Ldim,
      LH, 64, LH, 64, NH * LL, LL, NH * LL, LL, NH, 64, 0.125f, 0);
  k_softmax_bf16<<<(Bdim * NH * Ldim) / 8, 256, 0, stream>>>(scor, attnB);
  // out = attn @ V (B supplied [K][N] -> transpose on load)
  k_gemm<1><<<dim3(1, Ldim / 128, Bdim * NH), 256, 0, stream>>>(
      attnB, Vb, nullptr, aoB, nullptr, nullptr, Ldim, Hdim, Hdim, 0,
      NH * LL, LL, LH, 64, LH, 64, 0, 0, NH, Ldim, 1.0f, 0);
  // seq2 = seq + out @ Wo + bo
  k_gemm<0><<<dim3(Hdim / 64, Mtok / 128, 1), 256, 0, stream>>>(
      aoB, WoT, seq2, nullptr, seq, bo, Hdim, Hdim, Hdim, Hdim,
      0, 0, 0, 0, 0, 0, 0, 0, 1, Hdim, 1.0f, 0);
  // FFN
  k_ln_bf16<<<Mtok, 256, 0, stream>>>(seq2, ff_ln_g, ff_ln_b, xlnB, Hdim);
  k_gemm<0><<<dim3(FFdim / 64, Mtok / 128, 1), 256, 0, stream>>>(
      xlnB, W1T, nullptr, ffhB, nullptr, ff_b1, Hdim, Hdim, FFdim, 0,
      0, 0, 0, 0, 0, 0, 0, 0, 1, Hdim, 1.0f, 1);
  k_gemm<0><<<dim3(Hdim / 64, Mtok / 128, 1), 256, 0, stream>>>(
      ffhB, W2T, out_seq, soutB, seq2, ff_b2, FFdim, FFdim, Hdim, Hdim,
      0, 0, 0, 0, 0, 0, 0, 0, 1, FFdim, 1.0f, 0);
  // outer projections a = seq@Wi, b = seq@Wj
  k_gemm<0><<<dim3(Pdim / 64, Mtok / 128, 1), 256, 0, stream>>>(
      soutB, WoiT, aF, nullptr, nullptr, nullptr, Hdim, Hdim, Pdim, 0,
      0, 0, 0, 0, 0, 0, 0, 0, 1, Hdim, 1.0f, 0);
  k_gemm<0><<<dim3(Pdim / 64, Mtok / 128, 1), 256, 0, stream>>>(
      soutB, WojT, bF, nullptr, nullptr, nullptr, Hdim, Hdim, Pdim, 0,
      0, 0, 0, 0, 0, 0, 0, 0, 1, Hdim, 1.0f, 0);
  // fused pair transition
  k_pair_update<<<dim3(Ldim / 128, Ldim, Bdim), 256, 0, stream>>>(
      pair, aF, bF, outer_b, pu_ln_g, pu_ln_b, Wp1T, pu_b1, Wp2T, pu_b2,
      out_pair);
}
